// MultiHeadSelfAttention_36301063585895
// MI455X (gfx1250) — matrix operand order repaired and checked by reading, not compiled, not run on hardware
//
#include <hip/hip_runtime.h>
#include <hip/hip_bf16.h>
#include <math.h>

// ---------------- CDNA5 (gfx1250) wave32 WMMA types -------------------------
typedef __bf16 bf16;
typedef __attribute__((ext_vector_type(16))) __bf16 v16bf;
typedef __attribute__((ext_vector_type(8)))  float  v8f;

#define NDIM  1024
#define HEADS 16
#define HDIM  64
#define BATCH 8
#define SEQ   1024
#define MROWS (BATCH * SEQ)   // 8192

union Frag { v16bf v; uint4 u[2]; };

__device__ __forceinline__ v8f wmma_bf16(const v16bf& a, const v16bf& b, const v8f& c) {
  // (neg_a, A, neg_b, B, c_mod, C, reuse_a, reuse_b)
  return __builtin_amdgcn_wmma_f32_16x16x32_bf16(false, a, false, b, (short)0, c, false, false);
}

__device__ __forceinline__ v8f vzero8() {
  v8f z;
#pragma unroll
  for (int i = 0; i < 8; ++i) z[i] = 0.0f;
  return z;
}

// ---------------- f32 -> bf16 conversion (4 elems / thread) -----------------
__global__ __launch_bounds__(256) void cvt_bf16(const float* __restrict__ in,
                                                bf16* __restrict__ out, int n4) {
  int i = blockIdx.x * 256 + threadIdx.x;
  if (i < n4) {
    float4 v = ((const float4*)in)[i];
    union { bf16 h[4]; uint2 u; } pk;
    pk.h[0] = (bf16)v.x; pk.h[1] = (bf16)v.y; pk.h[2] = (bf16)v.z; pk.h[3] = (bf16)v.w;
    ((uint2*)out)[i] = pk.u;
  }
}

// ---------------- NT GEMM: C[m,n] = (sum_k A[m,k]*W[n,k] + bias[n])*oscale --
// A: bf16 [M x K] row-major ; W: bf16 [N x K] row-major (nn.Linear weight)
// MODE 0: out = bf16 [B,H,S,D]      (Q, K projections; Q folds 1/sqrt(D) here)
// MODE 1: out = bf16 [B,H,D,S]      (V projection, stored transposed)
// MODE 2: out = f32  [M,N]          (output projection -> d_out)
// Block: 256 threads = 8 waves (4 along M x 2 along N).
// Block tile 256(M) x 128(N); wave tile 64 x 64 -> 16 WMMAs / 16 B128 loads per K-step.
template <int MODE>
__global__ __launch_bounds__(256) void gemm_nt(const bf16* __restrict__ A,
                                               const bf16* __restrict__ W,
                                               const float* __restrict__ bias,
                                               void* __restrict__ out,
                                               int M, int N, int K, float oscale) {
  const int lane = threadIdx.x & 31;
  const int wave = threadIdx.x >> 5;
  const int m0 = blockIdx.y * 256 + (wave & 3) * 64;
  const int n0 = blockIdx.x * 128 + (wave >> 2) * 64;
  const int lrow = lane & 15;
  const int lhiA = (lane < 16) ? 0 : 16;   // A-fragment K split
  const int lhiB = (lane < 16) ? 0 : 16;   // B-fragment K split

  v8f acc[4][4];
#pragma unroll
  for (int i = 0; i < 4; ++i)
#pragma unroll
    for (int j = 0; j < 4; ++j) acc[i][j] = vzero8();

  for (int kc = 0; kc < K; kc += 32) {
    Frag a[4], b[4];
#pragma unroll
    for (int mi = 0; mi < 4; ++mi) {
      const uint4* p = (const uint4*)(A + (size_t)(m0 + mi * 16 + lrow) * K + kc + lhiA);
      a[mi].u[0] = p[0];   // k = kc+lhiA .. +7
      a[mi].u[1] = p[1];   // k = kc+lhiA+8 .. +15
    }
#pragma unroll
    for (int ni = 0; ni < 4; ++ni) {
      const uint4* p = (const uint4*)(W + (size_t)(n0 + ni * 16 + lrow) * K + kc + lhiB);
      b[ni].u[0] = p[0];   // k = kc+lhiB .. +7
      b[ni].u[1] = p[1];   // k = kc+lhiB+8 .. +15
    }
#pragma unroll
    for (int mi = 0; mi < 4; ++mi)
#pragma unroll
      for (int ni = 0; ni < 4; ++ni)
        acc[mi][ni] = wmma_bf16(a[mi].v, b[ni].v, acc[mi][ni]);
  }

  // C layout: VGPR r holds M = r (lanes 0-15) / r+8 (lanes 16-31), N = lane%16
  const int halfm = (lane < 16) ? 0 : 8;
#pragma unroll
  for (int mi = 0; mi < 4; ++mi) {
#pragma unroll
    for (int ni = 0; ni < 4; ++ni) {
#pragma unroll
      for (int r = 0; r < 8; ++r) {
        int m = m0 + mi * 16 + halfm + r;
        int n = n0 + ni * 16 + lrow;
        float val = (acc[mi][ni][r] + bias[n]) * oscale;
        if (MODE == 2) {
          ((float*)out)[(size_t)m * N + n] = val;
        } else {
          int bb = m >> 10, s = m & 1023;
          int h = n >> 6, d = n & 63;
          size_t idx = (MODE == 0)
                           ? (((size_t)bb * HEADS + h) * SEQ + s) * HDIM + d
                           : (((size_t)bb * HEADS + h) * HDIM + d) * SEQ + s;
          ((bf16*)out)[idx] = (bf16)val;
        }
      }
    }
  }
}

// ---------------- Flash attention: 1 wave per 32-query tile ------------------
// Q (pre-scaled by 1/sqrt(D)), K: bf16 [B,H,S,D] ; Vt: bf16 [B,H,D,S]
// mask: u8 [B,S] ; ctx: bf16 [B,S,NDIM]
// Row-sums of P via P @ ones on the matrix pipe; only row-max uses shuffles.
// exp(-inf)=0 semantics (with a finite clamp on the stabilizer) replace all
// -inf guard cndmasks; fully-masked rows finalize to 0 (== nan_to_num).
__global__ __launch_bounds__(256) void attn_kernel(const bf16* __restrict__ Q,
                                                   const bf16* __restrict__ Kk,
                                                   const bf16* __restrict__ Vt,
                                                   const unsigned char* __restrict__ mask,
                                                   bf16* __restrict__ ctx) {
  __shared__ __align__(16) bf16 lds_p[8][2][16 * 32];  // per-wave P tiles (2 x 16q x 32k)

  const int lane = threadIdx.x & 31;
  const int wave = threadIdx.x >> 5;
  const int gid = blockIdx.x * 8 + wave;  // 4096 tiles total
  const int qt = gid & 31;                // 32 query tiles (of 32) per (b,h)
  const int bh = gid >> 5;
  const int b = bh >> 4;
  const int h = bh & 15;
  const int q0 = qt * 32;
  const int lrow = lane & 15;
  const int lhiA = (lane < 16) ? 0 : 8;
  const int lhiB = (lane < 16) ? 0 : 16;

  // all-ones B fragment for row-sum WMMA
  Frag ones;
#pragma unroll
  for (int i = 0; i < 16; ++i) ones.v[i] = (bf16)1.0f;

  // Q fragments: [query subtile][d-chunk 0/1], A-layout
  Frag qa[2][2];
#pragma unroll
  for (int qs = 0; qs < 2; ++qs) {
    const bf16* qrow = Q + ((size_t)bh * SEQ + q0 + qs * 16 + lrow) * HDIM;
    const uint4* p0 = (const uint4*)(qrow + lhiB);
    qa[qs][0].u[0] = p0[0]; qa[qs][0].u[1] = p0[1];
    const uint4* p1 = (const uint4*)(qrow + 32 + lhiB);
    qa[qs][1].u[0] = p1[0]; qa[qs][1].u[1] = p1[1];
  }

  v8f acc[2][4], accl[2];
  float mrun[2][8];
#pragma unroll
  for (int qs = 0; qs < 2; ++qs) {
    accl[qs] = vzero8();
#pragma unroll
    for (int f = 0; f < 4; ++f) acc[qs][f] = vzero8();
#pragma unroll
    for (int r = 0; r < 8; ++r) mrun[qs][r] = -INFINITY;
  }

  for (int k0 = 0; k0 < SEQ; k0 += 32) {
    // ---- K fragments for two 16-key subtiles (shared by both query tiles) --
    Frag kb0[2], kb1[2];
#pragma unroll
    for (int dc = 0; dc < 2; ++dc) {
      const uint4* p0 = (const uint4*)(Kk + ((size_t)bh * SEQ + k0 + lrow) * HDIM + dc * 32 + lhiB);
      kb0[dc].u[0] = p0[0]; kb0[dc].u[1] = p0[1];
      const uint4* p1 = (const uint4*)(Kk + ((size_t)bh * SEQ + k0 + 16 + lrow) * HDIM + dc * 32 + lhiB);
      kb1[dc].u[0] = p1[0]; kb1[dc].u[1] = p1[1];
    }
    const bool msk0 = mask[(size_t)b * SEQ + k0 + lrow] != 0;
    const bool msk1 = mask[(size_t)b * SEQ + k0 + 16 + lrow] != 0;

#pragma unroll
    for (int qs = 0; qs < 2; ++qs) {
      v8f s0 = vzero8(), s1 = vzero8();
      s0 = wmma_bf16(qa[qs][0].v, kb0[0].v, s0);
      s0 = wmma_bf16(qa[qs][1].v, kb0[1].v, s0);
      s1 = wmma_bf16(qa[qs][0].v, kb1[0].v, s1);
      s1 = wmma_bf16(qa[qs][1].v, kb1[1].v, s1);

      // ---- online softmax (rows live across 16 lanes of each half) --------
#pragma unroll
      for (int r = 0; r < 8; ++r) {
        float x0 = msk0 ? -INFINITY : s0[r];
        float x1 = msk1 ? -INFINITY : s1[r];
        float mx = fmaxf(x0, x1);
#pragma unroll
        for (int off = 1; off < 16; off <<= 1) mx = fmaxf(mx, __shfl_xor(mx, off, 32));
        float mnew = fmaxf(mrun[qs][r], mx);
        float mc = fmaxf(mnew, -3.0e38f);   // finite stabilizer: exp(-inf - mc) == 0
        float corr = __expf(mrun[qs][r] - mc);
        float p0 = __expf(x0 - mc);
        float p1 = __expf(x1 - mc);
        mrun[qs][r] = mnew;
        accl[qs][r] *= corr;
#pragma unroll
        for (int f = 0; f < 4; ++f) acc[qs][f][r] *= corr;
        int row = r + lhiA;
        lds_p[wave][qs][row * 32 + lrow]      = (bf16)p0;
        lds_p[wave][qs][row * 32 + 16 + lrow] = (bf16)p1;
      }
    }

    asm volatile("s_wait_dscnt 0" ::: "memory");  // P store -> P reload fence

    // ---- reload P in A-layout; row-sum via P @ ones ------------------------
    Frag pa[2];
#pragma unroll
    for (int qs = 0; qs < 2; ++qs) {
      const uint4* p = (const uint4*)&lds_p[wave][qs][lrow * 32 + lhiB];
      pa[qs].u[0] = p[0];
      pa[qs].u[1] = p[1];
      accl[qs] = wmma_bf16(pa[qs].v, ones.v, accl[qs]);
    }

    // ---- P @ V: V fragments shared by both query tiles ---------------------
#pragma unroll
    for (int f = 0; f < 4; ++f) {
      Frag vb;
      const uint4* p = (const uint4*)(Vt + ((size_t)bh * HDIM + f * 16 + lrow) * SEQ + k0 + lhiB);
      vb.u[0] = p[0]; vb.u[1] = p[1];
#pragma unroll
      for (int qs = 0; qs < 2; ++qs)
        acc[qs][f] = wmma_bf16(pa[qs].v, vb.v, acc[qs][f]);
    }
  }

  // ---- finalize: ctx[b, s, h*64 + d] ; fully-masked rows -> 0 --------------
#pragma unroll
  for (int qs = 0; qs < 2; ++qs) {
#pragma unroll
    for (int r = 0; r < 8; ++r) {
      float l = accl[qs][r];  // every column of the row-sum WMMA holds sum_k P
      float inv = (l > 0.0f) ? 1.0f / l : 0.0f;
      int srow = q0 + qs * 16 + r + lhiA;
#pragma unroll
      for (int f = 0; f < 4; ++f) {
        float val = acc[qs][f][r] * inv;
        ctx[((size_t)b * SEQ + srow) * NDIM + h * HDIM + f * 16 + lrow] = (bf16)val;
      }
    }
  }
}

// ---------------- launch ----------------------------------------------------
extern "C" void kernel_launch(void* const* d_in, const int* in_sizes, int n_in,
                              void* d_out, int out_size, void* d_ws, size_t ws_size,
                              hipStream_t stream) {
  const float* x = (const float*)d_in[0];
  const unsigned char* mask = (const unsigned char*)d_in[1];
  const float* Wq = (const float*)d_in[2];
  const float* bq = (const float*)d_in[3];
  const float* Wk = (const float*)d_in[4];
  const float* bk = (const float*)d_in[5];
  const float* Wv = (const float*)d_in[6];
  const float* bv = (const float*)d_in[7];
  const float* Wo = (const float*)d_in[8];
  const float* bo = (const float*)d_in[9];
  float* out = (float*)d_out;

  char* ws = (char*)d_ws;
  constexpr size_t SZ_X = (size_t)MROWS * NDIM * sizeof(bf16);  // 16 MB
  constexpr size_t SZ_W = (size_t)NDIM * NDIM * sizeof(bf16);   // 2 MB
  bf16* xbf = (bf16*)(ws);
  bf16* Wqb = (bf16*)(ws + SZ_X);
  bf16* Wkb = (bf16*)(ws + SZ_X + SZ_W);
  bf16* Wvb = (bf16*)(ws + SZ_X + 2 * SZ_W);
  bf16* Wob = (bf16*)(ws + SZ_X + 3 * SZ_W);
  bf16* Qb  = (bf16*)(ws + SZ_X + 4 * SZ_W);
  bf16* Kb  = (bf16*)(ws + 2 * SZ_X + 4 * SZ_W);
  bf16* Vtb = (bf16*)(ws + 3 * SZ_X + 4 * SZ_W);
  bf16* ctxb = xbf;  // x is dead after the V projection -> reuse its buffer

  // f32 -> bf16
  {
    int n4 = MROWS * NDIM / 4;
    cvt_bf16<<<n4 / 256, 256, 0, stream>>>(x, xbf, n4);
    int w4 = NDIM * NDIM / 4;
    cvt_bf16<<<w4 / 256, 256, 0, stream>>>(Wq, Wqb, w4);
    cvt_bf16<<<w4 / 256, 256, 0, stream>>>(Wk, Wkb, w4);
    cvt_bf16<<<w4 / 256, 256, 0, stream>>>(Wv, Wvb, w4);
    cvt_bf16<<<w4 / 256, 256, 0, stream>>>(Wo, Wob, w4);
  }

  // Q/K/V projections (NT GEMM, bf16 WMMA, f32 accumulate)
  // Q folds the attention scale 1/sqrt(64) = 0.125 into its epilogue.
  dim3 ggrid(NDIM / 128, MROWS / 256);
  gemm_nt<0><<<ggrid, 256, 0, stream>>>(xbf, Wqb, bq, (void*)Qb,  MROWS, NDIM, NDIM, 0.125f);
  gemm_nt<0><<<ggrid, 256, 0, stream>>>(xbf, Wkb, bk, (void*)Kb,  MROWS, NDIM, NDIM, 1.0f);
  gemm_nt<1><<<ggrid, 256, 0, stream>>>(xbf, Wvb, bv, (void*)Vtb, MROWS, NDIM, NDIM, 1.0f);

  // attention: 4096 query tiles (32 queries each), 8 waves / block
  attn_kernel<<<(BATCH * HEADS * (SEQ / 32)) / 8, 256, 0, stream>>>(Qb, Kb, Vtb, mask, ctxb);

  // output projection -> f32 d_out
  gemm_nt<2><<<ggrid, 256, 0, stream>>>(ctxb, Wob, bo, (void*)out, MROWS, NDIM, NDIM, 1.0f);
}
